// MultiHeadAttention_76622216560988
// MI455X (gfx1250) — matrix operand order repaired and checked by reading, not compiled, not run on hardware
//
#include <hip/hip_runtime.h>
#include <hip/hip_bf16.h>
#include <math.h>

// ---------------------------------------------------------------- types ----
typedef __attribute__((ext_vector_type(16))) _Float16 v16h;
typedef __attribute__((ext_vector_type(8)))  _Float16 v8h;
typedef __attribute__((ext_vector_type(4)))  _Float16 v4h;
typedef __attribute__((ext_vector_type(8)))  float    v8f;

// problem dims
#define BB 2
#define LL 2048
#define EE 1024
#define HH 16
#define DH 64
#define NROW (BB * LL)        // 4096
#define NEG_BIG (-3.0e38f)

// ------------------------------------------------------- WMMA fragment IO --
__device__ __forceinline__ v16h pack16(v8h lo, v8h hi) {
  v16h r;
#pragma unroll
  for (int i = 0; i < 8; ++i) { r[i] = lo[i]; r[i + 8] = hi[i]; }
  return r;
}

// A-fragment (16x32 f16, row-major src). base points at (row0, k0).
__device__ __forceinline__ v16h load_frag_a(const _Float16* base, int ld, int lane) {
  const _Float16* p = base + (size_t)(lane & 15) * ld + (lane >> 4) * 8;
  v8h lo = *(const v8h*)p;          // K = khalf*8 .. +7
  v8h hi = *(const v8h*)(p + 16);   // K = 16 + khalf*8 .. +7
  return pack16(lo, hi);
}

// B-fragment (32x16). Source stored "output-column-major": row j of src is
// output column j, contiguous in K.  base points at (col0, k0).
__device__ __forceinline__ v16h load_frag_b(const _Float16* base, int ld, int lane) {
  const _Float16* p = base + (size_t)(lane & 15) * ld + (lane >> 4) * 8;
  v8h lo = *(const v8h*)p;          // K = khalf*8 .. +7
  v8h hi = *(const v8h*)(p + 16);   // K = 16 + khalf*8 .. +7
  return pack16(lo, hi);
}

__device__ __forceinline__ v8f wmma_f16(v16h a, v16h b, v8f c) {
  return __builtin_amdgcn_wmma_f32_16x16x32_f16(
      false, a, false, b, (short)0, c, false, false);
}

// ------------------------------------------------- kernel 1: f32 -> f16 ----
__global__ __launch_bounds__(256) void cvt_f16_kernel(
    const float* __restrict__ wq, const float* __restrict__ wk,
    const float* __restrict__ wv, const float* __restrict__ wo,
    const float* __restrict__ xq, const float* __restrict__ xk,
    const float* __restrict__ xv,
    _Float16* __restrict__ w16, _Float16* __restrict__ x16) {
  const int NW = 4 * EE * EE;
  const int NX = 3 * NROW * EE;
  int idx = blockIdx.x * blockDim.x + threadIdx.x;
  int i4 = idx * 4;
  if (i4 >= NW + NX) return;
  const float* src;
  _Float16* dst;
  if (i4 < NW) {
    int t = i4 >> 20, off = i4 & (EE * EE - 1);
    src = (t == 0) ? wq : (t == 1) ? wk : (t == 2) ? wv : wo;
    src += off;
    dst = w16 + (size_t)t * (EE * EE) + off;
  } else {
    int j = i4 - NW;
    int t = j >> 22, off = j & (NROW * EE - 1);
    src = (t == 0) ? xq : (t == 1) ? xk : xv;
    src += off;
    dst = x16 + (size_t)t * (NROW * EE) + off;
  }
  float4 v = *(const float4*)src;
  v4h h; h[0] = (_Float16)v.x; h[1] = (_Float16)v.y;
  h[2] = (_Float16)v.z; h[3] = (_Float16)v.w;
  *(v4h*)dst = h;
}

// ---------------- kernel 2: QKV projection, 32x32 tile per wave (WMMA) -----
// y = x @ W^T + b.  Writes Q,K head-major [b][h][l][d]; V transposed [b][h][d][l].
__global__ __launch_bounds__(256) void qkv_gemm_kernel(
    const _Float16* __restrict__ x16, const _Float16* __restrict__ w16,
    const float* __restrict__ bq, const float* __restrict__ bk,
    const float* __restrict__ bv,
    _Float16* __restrict__ qh, _Float16* __restrict__ kh,
    _Float16* __restrict__ vt) {
  int wave = (blockIdx.x * blockDim.x + threadIdx.x) >> 5;
  int lane = threadIdx.x & 31;
  int t   = wave >> 12;            // 4096 tiles per tensor
  int rem = wave & 4095;
  int mt  = rem >> 5;              // 128 row tiles of 32
  int nt  = rem & 31;              // 32  col tiles of 32
  const _Float16* A  = x16 + (size_t)t * (NROW * EE) + (size_t)(mt * 32) * EE;
  const _Float16* Bm = w16 + (size_t)t * (EE * EE) + (size_t)(nt * 32) * EE;
  const v8f vz = {0.f, 0.f, 0.f, 0.f, 0.f, 0.f, 0.f, 0.f};
  v8f acc[2][2] = {{vz, vz}, {vz, vz}};
#pragma unroll 2
  for (int k = 0; k < EE; k += 32) {
    __builtin_prefetch(A + k + 256, 0, 0);
    v16h a0 = load_frag_a(A + k, EE, lane);
    v16h a1 = load_frag_a(A + (size_t)16 * EE + k, EE, lane);
    v16h b0 = load_frag_b(Bm + k, EE, lane);
    v16h b1 = load_frag_b(Bm + (size_t)16 * EE + k, EE, lane);
    acc[0][0] = wmma_f16(a0, b0, acc[0][0]);
    acc[0][1] = wmma_f16(a0, b1, acc[0][1]);
    acc[1][0] = wmma_f16(a1, b0, acc[1][0]);
    acc[1][1] = wmma_f16(a1, b1, acc[1][1]);
  }
  const float* bias = (t == 0) ? bq : (t == 1) ? bk : bv;
  int cn = lane & 15, hf = lane >> 4;
#pragma unroll
  for (int tc = 0; tc < 2; ++tc) {
    int col = nt * 32 + tc * 16 + cn;
    float bval = bias[col];
    int h = col >> 6, d = col & 63;
#pragma unroll
    for (int tr = 0; tr < 2; ++tr) {
#pragma unroll
      for (int r = 0; r < 8; ++r) {
        int row = mt * 32 + tr * 16 + r + 8 * hf;
        int bi = row >> 11, l = row & (LL - 1);
        float val = acc[tr][tc][r] + bval;
        if (t == 2) {
          vt[(((size_t)bi * HH + h) * DH + d) * LL + l] = (_Float16)val;
        } else {
          _Float16* dst = (t == 0) ? qh : kh;
          dst[(((size_t)bi * HH + h) * LL + l) * DH + d] = (_Float16)val;
        }
      }
    }
  }
}

// ---------- kernel 3: flash attention, 32-query tile per wave (WMMA) -------
__global__ __launch_bounds__(256) void attn_kernel(
    const _Float16* __restrict__ qh, const _Float16* __restrict__ kh,
    const _Float16* __restrict__ vt, const unsigned char* __restrict__ mask,
    _Float16* __restrict__ ctx) {
  __shared__ __align__(16) _Float16 ps[8][2 * 16 * 32];  // per-wave P staging
  int wloc = threadIdx.x >> 5;
  int lane = threadIdx.x & 31;
  int w = blockIdx.x * 8 + wloc;            // 2048 waves total
  int b  = w >> 10;
  int rm = w & 1023;
  int h  = rm >> 6;
  int qt = rm & 63;                         // query tile (32 rows)
  int bh = b * HH + h;
  const _Float16* Q = qh + ((size_t)bh * LL + qt * 32) * DH;
  const _Float16* K = kh + (size_t)bh * LL * DH;
  const _Float16* V = vt + (size_t)bh * DH * LL;
  const unsigned char* M = mask + (size_t)b * LL * LL;

  // Q fragments: 2 row tiles x 2 k-steps, resident for the whole sweep
  v16h aq[2][2];
#pragma unroll
  for (int rt = 0; rt < 2; ++rt) {
    aq[rt][0] = load_frag_a(Q + (size_t)rt * 16 * DH, DH, lane);
    aq[rt][1] = load_frag_a(Q + (size_t)rt * 16 * DH + 32, DH, lane);
  }
  const v8f vz = {0.f, 0.f, 0.f, 0.f, 0.f, 0.f, 0.f, 0.f};
  v8f acc[2][4];
  float mrun[2][8], lrun[2][8];
#pragma unroll
  for (int rt = 0; rt < 2; ++rt) {
#pragma unroll
    for (int dt = 0; dt < 4; ++dt) acc[rt][dt] = vz;
#pragma unroll
    for (int r = 0; r < 8; ++r) { mrun[rt][r] = NEG_BIG; lrun[rt][r] = 0.f; }
  }

  int cn = lane & 15, hf = lane >> 4;
  _Float16* pw = &ps[wloc][0];

  for (int kc = 0; kc < LL / 32; ++kc) {
    int key0 = kc * 32;
    // K fragments (shared by both row tiles)
    v16h bk00 = load_frag_b(K + (size_t)key0 * DH, DH, lane);
    v16h bk01 = load_frag_b(K + (size_t)key0 * DH + 32, DH, lane);
    v16h bk10 = load_frag_b(K + (size_t)(key0 + 16) * DH, DH, lane);
    v16h bk11 = load_frag_b(K + (size_t)(key0 + 16) * DH + 32, DH, lane);
    // V fragments (shared by both row tiles)
    v16h bv0 = load_frag_b(V + (size_t)0 * 16 * LL + key0, LL, lane);
    v16h bv1 = load_frag_b(V + (size_t)1 * 16 * LL + key0, LL, lane);
    v16h bv2 = load_frag_b(V + (size_t)2 * 16 * LL + key0, LL, lane);
    v16h bv3 = load_frag_b(V + (size_t)3 * 16 * LL + key0, LL, lane);

#pragma unroll
    for (int rt = 0; rt < 2; ++rt) {
      v8f s0 = wmma_f16(aq[rt][0], bk00, vz); s0 = wmma_f16(aq[rt][1], bk01, s0);
      v8f s1 = wmma_f16(aq[rt][0], bk10, vz); s1 = wmma_f16(aq[rt][1], bk11, s1);

      int col0 = key0 + cn, col1 = key0 + 16 + cn;
      float p0[8], p1[8], alpha[8];
#pragma unroll
      for (int r = 0; r < 8; ++r) {
        float x0 = s0[r] * 0.125f;            // 1/sqrt(DH)
        float x1 = s1[r] * 0.125f;
        int lq = qt * 32 + rt * 16 + r + 8 * hf;
        if (M[(size_t)lq * LL + col0]) x0 = NEG_BIG;
        if (M[(size_t)lq * LL + col1]) x1 = NEG_BIG;
        float v = fmaxf(x0, x1);              // row max over 32 cols
        v = fmaxf(v, __shfl_xor(v, 1));
        v = fmaxf(v, __shfl_xor(v, 2));
        v = fmaxf(v, __shfl_xor(v, 4));
        v = fmaxf(v, __shfl_xor(v, 8));
        float mn = fmaxf(mrun[rt][r], v);
        float al = __expf(mrun[rt][r] - mn);
        mrun[rt][r] = mn; alpha[r] = al;
        p0[r] = __expf(x0 - mn);
        p1[r] = __expf(x1 - mn);
        float rs = p0[r] + p1[r];
        rs += __shfl_xor(rs, 1);
        rs += __shfl_xor(rs, 2);
        rs += __shfl_xor(rs, 4);
        rs += __shfl_xor(rs, 8);
        lrun[rt][r] = lrun[rt][r] * al + rs;
      }
#pragma unroll
      for (int r = 0; r < 8; ++r) {
        acc[rt][0][r] *= alpha[r]; acc[rt][1][r] *= alpha[r];
        acc[rt][2][r] *= alpha[r]; acc[rt][3][r] *= alpha[r];
      }
      // transpose P (C-layout -> A-layout) through LDS, intra-wave only
      _Float16* pt = pw + rt * 512;
#pragma unroll
      for (int r = 0; r < 8; ++r) {
        int row = r + 8 * hf;
        pt[row * 32 + cn]      = (_Float16)p0[r];
        pt[row * 32 + 16 + cn] = (_Float16)p1[r];
      }
      asm volatile("s_wait_dscnt 0" ::: "memory");  // wave-internal DS RAW
      v16h ap = load_frag_a(pt, 32, lane);
      acc[rt][0] = wmma_f16(ap, bv0, acc[rt][0]);
      acc[rt][1] = wmma_f16(ap, bv1, acc[rt][1]);
      acc[rt][2] = wmma_f16(ap, bv2, acc[rt][2]);
      acc[rt][3] = wmma_f16(ap, bv3, acc[rt][3]);
    }
  }

#pragma unroll
  for (int rt = 0; rt < 2; ++rt) {
#pragma unroll
    for (int r = 0; r < 8; ++r) {
      int lq = qt * 32 + rt * 16 + r + 8 * hf;
      size_t base = ((size_t)b * LL + lq) * EE + h * DH;
      float inv = 1.0f / lrun[rt][r];
      ctx[base + 0 * 16 + cn] = (_Float16)(acc[rt][0][r] * inv);
      ctx[base + 1 * 16 + cn] = (_Float16)(acc[rt][1][r] * inv);
      ctx[base + 2 * 16 + cn] = (_Float16)(acc[rt][2][r] * inv);
      ctx[base + 3 * 16 + cn] = (_Float16)(acc[rt][3][r] * inv);
    }
  }
}

// -------- kernel 4: output projection + bias + residual, 32x32/wave --------
__global__ __launch_bounds__(256) void out_gemm_kernel(
    const _Float16* __restrict__ ctx, const _Float16* __restrict__ wo16,
    const float* __restrict__ bo, const float* __restrict__ xq,
    float* __restrict__ outp) {
  int wave = (blockIdx.x * blockDim.x + threadIdx.x) >> 5;
  int lane = threadIdx.x & 31;
  int mt = wave >> 5;              // 128 row tiles of 32
  int nt = wave & 31;              // 32  col tiles of 32
  const _Float16* A  = ctx + (size_t)(mt * 32) * EE;
  const _Float16* Bm = wo16 + (size_t)(nt * 32) * EE;
  const v8f vz = {0.f, 0.f, 0.f, 0.f, 0.f, 0.f, 0.f, 0.f};
  v8f acc[2][2] = {{vz, vz}, {vz, vz}};
#pragma unroll 2
  for (int k = 0; k < EE; k += 32) {
    __builtin_prefetch(A + k + 256, 0, 0);
    v16h a0 = load_frag_a(A + k, EE, lane);
    v16h a1 = load_frag_a(A + (size_t)16 * EE + k, EE, lane);
    v16h b0 = load_frag_b(Bm + k, EE, lane);
    v16h b1 = load_frag_b(Bm + (size_t)16 * EE + k, EE, lane);
    acc[0][0] = wmma_f16(a0, b0, acc[0][0]);
    acc[0][1] = wmma_f16(a0, b1, acc[0][1]);
    acc[1][0] = wmma_f16(a1, b0, acc[1][0]);
    acc[1][1] = wmma_f16(a1, b1, acc[1][1]);
  }
  int cn = lane & 15, hf = lane >> 4;
#pragma unroll
  for (int tc = 0; tc < 2; ++tc) {
    int col = nt * 32 + tc * 16 + cn;
    float bval = bo[col];
#pragma unroll
    for (int tr = 0; tr < 2; ++tr) {
#pragma unroll
      for (int r = 0; r < 8; ++r) {
        int row = mt * 32 + tr * 16 + r + 8 * hf;
        size_t off = (size_t)row * EE + col;
        outp[off] = acc[tr][tc][r] + bval + xq[off];
      }
    }
  }
}

// ---------------------------- kernel 5: LayerNorm (block per row) ----------
__global__ __launch_bounds__(256) void ln_kernel(
    const float* __restrict__ outp, const float* __restrict__ gamma,
    const float* __restrict__ beta, float* __restrict__ out) {
  __shared__ float rs[256], rq[256];
  int row = blockIdx.x, tid = threadIdx.x;
  const float* x = outp + (size_t)row * EE;
  float4 v = *(const float4*)(x + tid * 4);
  float s = v.x + v.y + v.z + v.w;
  float q = v.x * v.x + v.y * v.y + v.z * v.z + v.w * v.w;
  rs[tid] = s; rq[tid] = q;
  __syncthreads();
  for (int off = 128; off > 0; off >>= 1) {
    if (tid < off) { rs[tid] += rs[tid + off]; rq[tid] += rq[tid + off]; }
    __syncthreads();
  }
  float mu = rs[0] * (1.0f / EE);
  float var = rq[0] * (1.0f / EE) - mu * mu;
  float rstd = rsqrtf(var + 1e-6f);
  int c = tid * 4;
  float4 g = *(const float4*)(gamma + c);
  float4 bb = *(const float4*)(beta + c);
  float4 o;
  o.x = (v.x - mu) * rstd * g.x + bb.x;
  o.y = (v.y - mu) * rstd * g.y + bb.y;
  o.z = (v.z - mu) * rstd * g.z + bb.z;
  o.w = (v.w - mu) * rstd * g.w + bb.w;
  *(float4*)(out + (size_t)row * EE + c) = o;
}

// ------------------------------------------------------------- launcher ----
extern "C" void kernel_launch(void* const* d_in, const int* in_sizes, int n_in,
                              void* d_out, int out_size, void* d_ws, size_t ws_size,
                              hipStream_t stream) {
  const float* xq = (const float*)d_in[0];
  const float* xk = (const float*)d_in[1];
  const float* xv = (const float*)d_in[2];
  const unsigned char* mask = (const unsigned char*)d_in[3];
  const float* Wq = (const float*)d_in[4];
  const float* bq = (const float*)d_in[5];
  const float* Wk = (const float*)d_in[6];
  const float* bk = (const float*)d_in[7];
  const float* Wv = (const float*)d_in[8];
  const float* bv = (const float*)d_in[9];
  const float* Wo = (const float*)d_in[10];
  const float* bo = (const float*)d_in[11];
  const float* gamma = (const float*)d_in[12];
  const float* beta  = (const float*)d_in[13];
  float* out = (float*)d_out;

  // workspace carve (80 MB total, all 16B-aligned)
  char* p = (char*)d_ws;
  _Float16* w16 = (_Float16*)p; p += (size_t)4 * EE * EE * 2;        //  8 MB
  _Float16* x16 = (_Float16*)p; p += (size_t)3 * NROW * EE * 2;      // 24 MB
  _Float16* qh  = (_Float16*)p; p += (size_t)NROW * EE * 2;          //  8 MB
  _Float16* kh  = (_Float16*)p; p += (size_t)NROW * EE * 2;          //  8 MB
  _Float16* vt  = (_Float16*)p; p += (size_t)NROW * EE * 2;          //  8 MB
  _Float16* ctx = (_Float16*)p; p += (size_t)NROW * EE * 2;          //  8 MB
  float*    outp = (float*)p;                                        // 16 MB

  cvt_f16_kernel<<<16384, 256, 0, stream>>>(Wq, Wk, Wv, Wo, xq, xk, xv, w16, x16);
  qkv_gemm_kernel<<<1536, 256, 0, stream>>>(x16, w16, bq, bk, bv, qh, kh, vt);
  attn_kernel<<<256, 256, 0, stream>>>(qh, kh, vt, mask, ctx);
  out_gemm_kernel<<<512, 256, 0, stream>>>(ctx, w16 + (size_t)3 * EE * EE,
                                           bo, xq, outp);
  ln_kernel<<<NROW, 256, 0, stream>>>(outp, gamma, beta, out);
}
